// GPTModel_75333726372074
// MI455X (gfx1250) — matrix operand order repaired and checked by reading, not compiled, not run on hardware
//
#include <hip/hip_runtime.h>

typedef __attribute__((ext_vector_type(16))) __bf16 v16bf;
typedef __attribute__((ext_vector_type(8)))  float  v8f;
typedef __attribute__((ext_vector_type(4)))  __bf16 v4bf;

#define GPT_D   768
#define GPT_S   1024
#define GPT_B   4
#define GPT_H   12
#define GPT_HD  64
#define GPT_NL  12
#define GPT_V   50257
#define GPT_M   (GPT_B * GPT_S)   // 4096 rows of the activation matrix

__device__ __forceinline__ v8f zero8() {
    v8f z = {0.f, 0.f, 0.f, 0.f, 0.f, 0.f, 0.f, 0.f};
    return z;
}

__device__ __forceinline__ float gelu_tanh(float x) {
    float x3 = x * x * x;
    return 0.5f * x * (1.0f + tanhf(0.7978845608028654f * (x + 0.044715f * x3)));
}

// gfx1250 async copy: global -> LDS, 16 bytes per lane, tracked by ASYNCcnt.
__device__ __forceinline__ void async_copy_b128(uint32_t lds_off, uint64_t gaddr) {
    asm volatile("global_load_async_to_lds_b128 %0, %1, off"
                 :: "v"(lds_off), "v"(gaddr) : "memory");
}
__device__ __forceinline__ void wait_asynccnt0() {
    asm volatile("s_wait_asynccnt 0" ::: "memory");
}

// ---------------------------------------------------------------------------
// x = tok_emb[idx] + pos_emb  (fp32 residual stream)
// ---------------------------------------------------------------------------
__global__ __launch_bounds__(256) void gpt_embed(const int* __restrict__ idx,
                                                 const float* __restrict__ tok,
                                                 const float* __restrict__ pos,
                                                 float* __restrict__ x) {
    int i = blockIdx.x * 256 + threadIdx.x;          // grid covers exactly M*D
    int row = i / GPT_D;
    int col = i - row * GPT_D;
    int t = idx[row];
    x[i] = tok[(size_t)t * GPT_D + col] + pos[(size_t)(row & (GPT_S - 1)) * GPT_D + col];
}

// ---------------------------------------------------------------------------
// LayerNorm (fp32 in) -> bf16 out.  Faithful to ref: eps NOT added.
// ---------------------------------------------------------------------------
__global__ __launch_bounds__(256) void gpt_layernorm(const float* __restrict__ x,
                                                     const float* __restrict__ sc,
                                                     const float* __restrict__ sh,
                                                     __bf16* __restrict__ out) {
    __shared__ float rsum[256];
    __shared__ float rsq[256];
    int row = blockIdx.x;
    int t = threadIdx.x;
    const float* xr = x + (size_t)row * GPT_D;
    float a0 = xr[t], a1 = xr[t + 256], a2 = xr[t + 512];
    rsum[t] = a0 + a1 + a2;
    rsq[t]  = a0 * a0 + a1 * a1 + a2 * a2;
    __syncthreads();
    for (int off = 128; off > 0; off >>= 1) {
        if (t < off) { rsum[t] += rsum[t + off]; rsq[t] += rsq[t + off]; }
        __syncthreads();
    }
    float mean = rsum[0] * (1.0f / GPT_D);
    float var  = rsq[0] * (1.0f / GPT_D) - mean * mean;
    float rstd = rsqrtf(var);
    __bf16* orow = out + (size_t)row * GPT_D;
#pragma unroll
    for (int j = 0; j < 3; ++j) {
        int col = t + j * 256;
        float v = (xr[col] - mean) * rstd * sc[col] + sh[col];
        orow[col] = (__bf16)v;
    }
}

// ---------------------------------------------------------------------------
// Tiled WMMA GEMM:  C[M,N] = act( A_bf16[M,K] * B_f32->bf16[K,N] + bias ) (+res)
// Block = 256 threads = 8 waves.  128x64 block tile; each wave owns 32x32
// (2x2 accumulators, A-fragment reused across both B-fragments).
// Double-buffered LDS; A tile staged with GLOBAL_LOAD_ASYNC_TO_LDS_B128,
// B tile converted fp32->bf16 and stored K-transposed.
// K multiple of 32; M multiple of 128; N guarded (logits N=50257).
// ---------------------------------------------------------------------------
template <bool GELU, bool RES, bool OUTBF16>
__global__ __launch_bounds__(256) void gpt_gemm(const __bf16* __restrict__ A,
                                                const float* __restrict__ B,
                                                const float* __restrict__ bias,
                                                void* __restrict__ Cout,
                                                int M, int N, int K) {
    __shared__ __bf16 As[2][128][40];   // [m][k], 80B row stride (16B aligned)
    __shared__ __bf16 BsT[2][64][36];   // [n][k] (transposed)

    const int tid  = threadIdx.x;
    const int lane = tid & 31;
    const int wave = tid >> 5;
    const int wm   = wave & 3;          // 4 waves along M (32 rows each)
    const int wn   = wave >> 2;         // 2 waves along N (32 cols each)
    const int ln   = lane & 15;
    const int kh   = lane >> 4;
    const int m0   = blockIdx.y * 128;
    const int n0   = blockIdx.x * 64;

    v8f acc[2][2];
#pragma unroll
    for (int ma = 0; ma < 2; ++ma)
#pragma unroll
        for (int nt = 0; nt < 2; ++nt) acc[ma][nt] = zero8();

    const int steps = K >> 5;

    auto stage = [&](int s, int buf) {
        const int kk = s << 5;
        // A tile: 128x32 bf16 = 8KB, async B128, 2 issues per thread
        for (int c = tid; c < 512; c += 256) {
            int row = c >> 2;
            int k8  = (c & 3) << 3;
            uint64_t ga = (uint64_t)(uintptr_t)(A + (size_t)(m0 + row) * K + kk + k8);
            uint32_t la = (uint32_t)(uintptr_t)&As[buf][row][k8];
            async_copy_b128(la, ga);
        }
        // B tile: 32x64 fp32 -> bf16, transposed to [n][k]
        for (int c = tid; c < 512; c += 256) {
            int kr = c >> 4;
            int n4 = (c & 15) << 2;
            const float* src = B + (size_t)(kk + kr) * N + n0 + n4;
#pragma unroll
            for (int i = 0; i < 4; ++i) {
                float v = (n0 + n4 + i < N) ? src[i] : 0.0f;
                BsT[buf][n4 + i][kr] = (__bf16)v;
            }
        }
    };

    stage(0, 0);
    wait_asynccnt0();
    __syncthreads();

    for (int s = 0; s < steps; ++s) {
        const int buf = s & 1;
        if (s + 1 < steps) stage(s + 1, buf ^ 1);   // overlap with compute

        // --- A fragments: 16x32, contiguous-K layout (M=lane%16; K half-split) ---
        v16bf af[2];
#pragma unroll
        for (int ma = 0; ma < 2; ++ma) {
            int am = wm * 32 + ma * 16 + ln;
#pragma unroll
            for (int i = 0; i < 8; ++i) {
                int kb = 2 * i + 16 * kh;
                af[ma][2 * i]     = As[buf][am][kb];
                af[ma][2 * i + 1] = As[buf][am][kb + 1];
            }
        }
        // --- B fragments: 32x16 each, N=lane%16, K = 16*(lane/16)+2i+j ---
        int bn = wn * 32 + ln;
        v16bf bfr[2];
#pragma unroll
        for (int nt = 0; nt < 2; ++nt) {
#pragma unroll
            for (int i = 0; i < 8; ++i) {
                int kb = 16 * kh + 2 * i;
                bfr[nt][2 * i]     = BsT[buf][bn + nt * 16][kb];
                bfr[nt][2 * i + 1] = BsT[buf][bn + nt * 16][kb + 1];
            }
        }
#pragma unroll
        for (int ma = 0; ma < 2; ++ma)
#pragma unroll
            for (int nt = 0; nt < 2; ++nt)
                acc[ma][nt] = __builtin_amdgcn_wmma_f32_16x16x32_bf16(
                    false, af[ma], false, bfr[nt], (short)0, acc[ma][nt], false, false);

        wait_asynccnt0();
        __syncthreads();
    }

    // --- epilogue: C layout N=lane%16, M = 8*(lane/16)+r ---
#pragma unroll
    for (int ma = 0; ma < 2; ++ma) {
#pragma unroll
        for (int nt = 0; nt < 2; ++nt) {
            int n = n0 + wn * 32 + nt * 16 + ln;
            if (n >= N) continue;
            float bv = bias ? bias[n] : 0.0f;
#pragma unroll
            for (int r = 0; r < 8; ++r) {
                int m = m0 + wm * 32 + ma * 16 + 8 * kh + r;
                float v = acc[ma][nt][r] + bv;
                if (GELU) v = gelu_tanh(v);
                size_t o = (size_t)m * N + n;
                if (OUTBF16) {
                    ((__bf16*)Cout)[o] = (__bf16)v;
                } else {
                    float* C = (float*)Cout;
                    if (RES) v += C[o];
                    C[o] = v;
                }
            }
        }
    }
}

// ---------------------------------------------------------------------------
// Flash attention (causal, scale 1/8).  Block = 128 threads = 4 waves; each
// wave owns a 16-row query tile; block shares 32-key K/V tiles through LDS.
// K tile staged with async-to-LDS; V transposed manually.  Online softmax
// with 16-lane shfl_xor row reductions (wave32 halves).
// q,k,v,ctx are bf16 [4096, 768] with column = h*64 + hd.
// ---------------------------------------------------------------------------
__global__ __launch_bounds__(128) void gpt_attention(const __bf16* __restrict__ q,
                                                     const __bf16* __restrict__ k,
                                                     const __bf16* __restrict__ v,
                                                     __bf16* __restrict__ ctx) {
    __shared__ __bf16 Kt[32][72];    // [key][hd], 144B row stride (16B aligned)
    __shared__ __bf16 VtT[64][36];   // [hd][key]
    __shared__ __bf16 Pt[4][16][32]; // per-wave P transpose staging

    const int tid  = threadIdx.x;
    const int lane = tid & 31;
    const int wave = tid >> 5;
    const int ln   = lane & 15;
    const int kh   = lane >> 4;

    const int qb  = blockIdx.x & 15;      // S/64 = 16 query super-tiles
    const int bh  = blockIdx.x >> 4;
    const int h   = bh % GPT_H;
    const int b   = bh / GPT_H;
    const int q0b = qb * 64;
    const int q0  = q0b + wave * 16;

    const size_t base = ((size_t)b * GPT_S) * GPT_D + (size_t)h * GPT_HD;

    // preload Q fragments (two 32-wide hd chunks), A layout
    v16bf qf[2];
    {
        const __bf16* qrow = q + base + (size_t)(q0 + ln) * GPT_D;
#pragma unroll
        for (int kc = 0; kc < 2; ++kc)
#pragma unroll
            for (int i = 0; i < 8; ++i) {
                int kb = kc * 32 + 2 * i + 16 * kh;
                qf[kc][2 * i]     = qrow[kb];
                qf[kc][2 * i + 1] = qrow[kb + 1];
            }
    }

    float mrow[8], lrow[8];
    v8f oacc[4];
#pragma unroll
    for (int r = 0; r < 8; ++r) { mrow[r] = -3.0e38f; lrow[r] = 0.0f; }
#pragma unroll
    for (int c = 0; c < 4; ++c) oacc[c] = zero8();

    const float scale = 0.125f;       // 1/sqrt(64)
    const int numj = (q0b >> 5) + 2;  // key blocks for causal coverage

    for (int j = 0; j < numj; ++j) {
        int kj = j * 32;
        __syncthreads();
        // stage K tile [32 keys][64 hd] via async-to-LDS (2 B128 per thread)
        for (int c = tid; c < 256; c += 128) {
            int key = c >> 3;
            int h8  = (c & 7) << 3;
            uint64_t ga = (uint64_t)(uintptr_t)(k + base + (size_t)(kj + key) * GPT_D + h8);
            uint32_t la = (uint32_t)(uintptr_t)&Kt[key][h8];
            async_copy_b128(la, ga);
        }
        // stage V tile transposed [64 hd][32 keys]
        for (int c = tid; c < 512; c += 128) {
            int key = c >> 4;
            int hd4 = (c & 15) << 2;
            const v4bf* src = (const v4bf*)(v + base + (size_t)(kj + key) * GPT_D + hd4);
            v4bf vv = *src;
#pragma unroll
            for (int i = 0; i < 4; ++i) VtT[hd4 + i][key] = vv[i];
        }
        wait_asynccnt0();
        __syncthreads();

        // S = Q * K^T  (16 x 32 scores: 2 accumulators x 2 hd chunks)
        v8f sacc[2];
        sacc[0] = zero8();
        sacc[1] = zero8();
#pragma unroll
        for (int nt = 0; nt < 2; ++nt) {
            int keyt = nt * 16 + ln;
#pragma unroll
            for (int kc = 0; kc < 2; ++kc) {
                v16bf kf;
#pragma unroll
                for (int i = 0; i < 8; ++i) {
                    int hd = kc * 32 + 16 * kh + 2 * i;
                    kf[2 * i]     = Kt[keyt][hd];
                    kf[2 * i + 1] = Kt[keyt][hd + 1];
                }
                sacc[nt] = __builtin_amdgcn_wmma_f32_16x16x32_bf16(false, qf[kc], false, kf,
                                                                   (short)0, sacc[nt], false, false);
            }
        }

        // online softmax (row stats across 16 lanes of each half-wave)
        float p[2][8];
#pragma unroll
        for (int r = 0; r < 8; ++r) {
            int row = q0 + 8 * kh + r;
#pragma unroll
            for (int nt = 0; nt < 2; ++nt) {
                int key = kj + nt * 16 + ln;
                float s = sacc[nt][r] * scale;
                p[nt][r] = (key <= row) ? s : -1.0e30f;
            }
        }
#pragma unroll
        for (int r = 0; r < 8; ++r) {
            float mb = fmaxf(p[0][r], p[1][r]);
#pragma unroll
            for (int off = 8; off >= 1; off >>= 1)
                mb = fmaxf(mb, __shfl_xor(mb, off, 32));
            float mn = fmaxf(mrow[r], mb);
            float sc = __expf(mrow[r] - mn);
            float p0 = __expf(p[0][r] - mn);
            float p1 = __expf(p[1][r] - mn);
            p[0][r] = p0;
            p[1][r] = p1;
            float rs = p0 + p1;
#pragma unroll
            for (int off = 8; off >= 1; off >>= 1)
                rs += __shfl_xor(rs, off, 32);
            lrow[r] = lrow[r] * sc + rs;
            mrow[r] = mn;
#pragma unroll
            for (int c = 0; c < 4; ++c) oacc[c][r] *= sc;
        }

        // transpose P (C layout -> A layout) through wave-private LDS
#pragma unroll
        for (int r = 0; r < 8; ++r)
#pragma unroll
            for (int nt = 0; nt < 2; ++nt)
                Pt[wave][8 * kh + r][nt * 16 + ln] = (__bf16)p[nt][r];

        v16bf pf;  // A-layout fragment over 32 keys (LDS in-order per wave)
#pragma unroll
        for (int i = 0; i < 8; ++i) {
            int kb = 2 * i + 16 * kh;
            pf[2 * i]     = Pt[wave][ln][kb];
            pf[2 * i + 1] = Pt[wave][ln][kb + 1];
        }

        // O += P * V  (4 n-subtiles of 16 hd columns)
#pragma unroll
        for (int c = 0; c < 4; ++c) {
            v16bf vf;
#pragma unroll
            for (int i = 0; i < 8; ++i) {
                int kb = 16 * kh + 2 * i;
                vf[2 * i]     = VtT[c * 16 + ln][kb];
                vf[2 * i + 1] = VtT[c * 16 + ln][kb + 1];
            }
            oacc[c] = __builtin_amdgcn_wmma_f32_16x16x32_bf16(false, pf, false, vf,
                                                              (short)0, oacc[c], false, false);
        }
    }

    // normalize and store ctx (bf16)
#pragma unroll
    for (int r = 0; r < 8; ++r) {
        float ol = 1.0f / lrow[r];
        size_t row = (size_t)(q0 + 8 * kh + r);
#pragma unroll
        for (int c = 0; c < 4; ++c)
            ctx[base + row * GPT_D + c * 16 + ln] = (__bf16)(oacc[c][r] * ol);
    }
}

// ---------------------------------------------------------------------------
// Host-side orchestration
// ---------------------------------------------------------------------------
extern "C" void kernel_launch(void* const* d_in, const int* in_sizes, int n_in,
                              void* d_out, int out_size, void* d_ws, size_t ws_size,
                              hipStream_t stream) {
    const int*   in_idx  = (const int*)d_in[0];
    const float* tok_emb = (const float*)d_in[1];
    const float* pos_emb = (const float*)d_in[2];
    const float* wq      = (const float*)d_in[3];
    const float* wk      = (const float*)d_in[4];
    const float* wv      = (const float*)d_in[5];
    const float* wo      = (const float*)d_in[6];
    const float* bo      = (const float*)d_in[7];
    const float* w1      = (const float*)d_in[8];
    const float* b1      = (const float*)d_in[9];
    const float* w2      = (const float*)d_in[10];
    const float* b2      = (const float*)d_in[11];
    const float* ln1_s   = (const float*)d_in[12];
    const float* ln1_b   = (const float*)d_in[13];
    const float* ln2_s   = (const float*)d_in[14];
    const float* ln2_b   = (const float*)d_in[15];
    const float* lnf_s   = (const float*)d_in[16];
    const float* lnf_b   = (const float*)d_in[17];
    const float* out_w   = (const float*)d_in[18];

    // workspace layout (bytes)
    const size_t SZ_X  = (size_t)GPT_M * GPT_D * sizeof(float);   // 12.6 MB
    const size_t SZ_BF = (size_t)GPT_M * GPT_D * sizeof(__bf16);  // 6.3 MB
    char* ws = (char*)d_ws;
    float*  x    = (float*)ws;
    __bf16* h    = (__bf16*)(ws + SZ_X);
    __bf16* qbuf = (__bf16*)(ws + SZ_X + SZ_BF);
    __bf16* kbuf = (__bf16*)(ws + SZ_X + 2 * SZ_BF);
    __bf16* vbuf = (__bf16*)(ws + SZ_X + 3 * SZ_BF);
    __bf16* cbuf = (__bf16*)(ws + SZ_X + 4 * SZ_BF);
    __bf16* fbuf = (__bf16*)(ws + SZ_X + 5 * SZ_BF);
    (void)in_sizes; (void)n_in; (void)out_size; (void)ws_size;

    const dim3 blk256(256), blk128(128);
    const dim3 gEmbed((GPT_M * GPT_D) / 256);
    const dim3 gLN(GPT_M);
    const dim3 gDD(GPT_D / 64, GPT_M / 128);          // 12 x 32
    const dim3 gDF((4 * GPT_D) / 64, GPT_M / 128);    // 48 x 32
    const dim3 gAttn(GPT_B * GPT_H * (GPT_S / 64));   // 768
    const dim3 gLog((GPT_V + 63) / 64, GPT_M / 128);  // 786 x 32

    gpt_embed<<<gEmbed, blk256, 0, stream>>>(in_idx, tok_emb, pos_emb, x);

    for (int l = 0; l < GPT_NL; ++l) {
        const size_t wdd = (size_t)l * GPT_D * GPT_D;
        const size_t wdf = (size_t)l * GPT_D * 4 * GPT_D;
        const size_t vd  = (size_t)l * GPT_D;
        const size_t vf  = (size_t)l * 4 * GPT_D;

        gpt_layernorm<<<gLN, blk256, 0, stream>>>(x, ln1_s + vd, ln1_b + vd, h);

        gpt_gemm<false, false, true><<<gDD, blk256, 0, stream>>>(
            h, wq + wdd, nullptr, qbuf, GPT_M, GPT_D, GPT_D);
        gpt_gemm<false, false, true><<<gDD, blk256, 0, stream>>>(
            h, wk + wdd, nullptr, kbuf, GPT_M, GPT_D, GPT_D);
        gpt_gemm<false, false, true><<<gDD, blk256, 0, stream>>>(
            h, wv + wdd, nullptr, vbuf, GPT_M, GPT_D, GPT_D);

        gpt_attention<<<gAttn, blk128, 0, stream>>>(qbuf, kbuf, vbuf, cbuf);

        gpt_gemm<false, true, false><<<gDD, blk256, 0, stream>>>(
            cbuf, wo + wdd, bo + vd, x, GPT_M, GPT_D, GPT_D);

        gpt_layernorm<<<gLN, blk256, 0, stream>>>(x, ln2_s + vd, ln2_b + vd, h);

        gpt_gemm<true, false, true><<<gDF, blk256, 0, stream>>>(
            h, w1 + wdf, b1 + vf, fbuf, GPT_M, 4 * GPT_D, GPT_D);
        gpt_gemm<false, true, false><<<gDD, blk256, 0, stream>>>(
            fbuf, w2 + wdf, b2 + vd, x, GPT_M, GPT_D, 4 * GPT_D);
    }

    gpt_layernorm<<<gLN, blk256, 0, stream>>>(x, lnf_s, lnf_b, h);
    gpt_gemm<false, false, false><<<gLog, blk256, 0, stream>>>(
        h, out_w, nullptr, d_out, GPT_M, GPT_V, GPT_D);
}
